// SAGD_75814762709185
// MI455X (gfx1250) — matrix operand order repaired and checked by reading, not compiled, not run on hardware
//
#include <hip/hip_runtime.h>
#include <hip/hip_bf16.h>

// ---------------------------------------------------------------------------
// Problem constants (match reference)
// ---------------------------------------------------------------------------
#define NN      50000     // NOTE: 50000 % 16 == 0 -> all row tiles are full
#define F_IN    256
#define HDIM    64
#define KDIM    32

typedef __attribute__((ext_vector_type(16))) __bf16 v16bf;
typedef __attribute__((ext_vector_type(8)))  __bf16 v8bf;
typedef __attribute__((ext_vector_type(8)))  float  v8f;

// ---------------------------------------------------------------------------
// Helpers
// ---------------------------------------------------------------------------
__device__ __forceinline__ void atomAddF(float* p, float v) {
    __hip_atomic_fetch_add(p, v, __ATOMIC_RELAXED, __HIP_MEMORY_SCOPE_AGENT);
}

// Monotonic float -> uint key so atomicMax(uint) == float max.
__device__ __forceinline__ unsigned fkey(float f) {
    unsigned b = __float_as_uint(f);
    return (b & 0x80000000u) ? ~b : (b | 0x80000000u);
}
__device__ __forceinline__ float funkey(unsigned k) {
    unsigned b = (k & 0x80000000u) ? (k ^ 0x80000000u) : ~k;
    return __uint_as_float(b);
}

__device__ __forceinline__ v8f wmma_bf16(v16bf a, v16bf b, v8f c) {
    // (neg_a, A, neg_b, B, c_mod, C, reuse_a, reuse_b)
    return __builtin_amdgcn_wmma_f32_16x16x32_bf16(false, a, false, b,
                                                   (short)0, c, false, false);
}

// ---------------------------------------------------------------------------
// bf16 WMMA GEMM:  C[M x Ncols] = cvt_bf16(A[M x Kd]) @ cvt_bf16(B[Kd x Ncols])
// Row-major inputs. Requires: M % 16 == 0, Kd % 32 == 0, Ncols % (16*CT) == 0.
//
// The whole B panel is staged once per block into LDS as a column-major bf16
// panel (ReLU folded in), padded to ldb = Kd+8 elements so half-wave 128-bit
// reads hit all 64 banks conflict-free. Each wave computes 16 rows and reuses
// one A fragment across CT WMMAs per K step.
// ---------------------------------------------------------------------------
template<int CT, bool RELU>
__global__ void gemm_wmma(const float* __restrict__ A,
                          const float* __restrict__ B,
                          float* __restrict__ C,
                          int M, int Kd, int Ncols)
{
    extern __shared__ __bf16 sB[];          // [Ncols][Kd+8], col-major
    const int tid  = threadIdx.x;
    const int lane = tid & 31;
    const int ldb  = Kd + 8;

    // Cooperative fill: sB[n][k] = cvt_bf16(relu?(B[k][n]))  (coalesced reads)
    for (int idx = tid; idx < Kd * Ncols; idx += blockDim.x) {
        int k = idx / Ncols;
        int n = idx - k * Ncols;
        float bv = B[idx];
        if (RELU) bv = bv > 0.f ? bv : 0.f;
        sB[(size_t)n * ldb + k] = (__bf16)bv;
    }
    __syncthreads();

    const int rowTile = blockIdx.x * (blockDim.x >> 5) + (tid >> 5);
    const int row0    = rowTile * 16;
    if (row0 >= M) return;

    const int half = lane >> 4;    // 0: lanes 0-15, 1: lanes 16-31
    const int l16  = lane & 15;
    const float* Arow = A + (long)(row0 + l16) * Kd;   // M%16==0: always valid

    for (int cg = 0; cg < Ncols; cg += 16 * CT) {
        v8f acc[CT];
#pragma unroll
        for (int t = 0; t < CT; ++t) acc[t] = (v8f){};

        for (int k0 = 0; k0 < Kd; k0 += 32) {
            // A: 16-bit 16x32 fragment. lane<16 holds K={0..7,16..23},
            // lane>=16 holds K={8..15,24..31} of row M=l16.
            // Chunks are 32B-aligned -> 4x global_load_b128.
            const float4 a0 = *(const float4*)(Arow + k0 + 8 * half);
            const float4 a1 = *(const float4*)(Arow + k0 + 8 * half + 4);
            const float4 a2 = *(const float4*)(Arow + k0 + 16 + 8 * half);
            const float4 a3 = *(const float4*)(Arow + k0 + 16 + 8 * half + 4);
            v16bf af;
            af[0]  = (__bf16)a0.x; af[1]  = (__bf16)a0.y;
            af[2]  = (__bf16)a0.z; af[3]  = (__bf16)a0.w;
            af[4]  = (__bf16)a1.x; af[5]  = (__bf16)a1.y;
            af[6]  = (__bf16)a1.z; af[7]  = (__bf16)a1.w;
            af[8]  = (__bf16)a2.x; af[9]  = (__bf16)a2.y;
            af[10] = (__bf16)a2.z; af[11] = (__bf16)a2.w;
            af[12] = (__bf16)a3.x; af[13] = (__bf16)a3.y;
            af[14] = (__bf16)a3.z; af[15] = (__bf16)a3.w;

#pragma unroll
            for (int t = 0; t < CT; ++t) {
                // B: 32x16 (KxN). Lane group selects K half, l16 = column.
                // K={8h..8h+7, 16+8h..16+8h+7} in LDS -> 2x ds_load_b128.
                const __bf16* p = sB + (size_t)(cg + 16 * t + l16) * ldb
                                     + k0 + 8 * half;
                v8bf blo = *(const v8bf*)(p);
                v8bf bhi = *(const v8bf*)(p + 16);
                v16bf bf = __builtin_shufflevector(
                    blo, bhi, 0, 1, 2, 3, 4, 5, 6, 7,
                              8, 9, 10, 11, 12, 13, 14, 15);
                acc[t] = wmma_bf16(af, bf, acc[t]);
            }
        }
        // C/D layout: VGPR r holds M = r (lanes 0-15) or M = r+8 (lanes 16-31)
#pragma unroll
        for (int t = 0; t < CT; ++t) {
#pragma unroll
            for (int r = 0; r < 8; ++r) {
                int row = row0 + r + 8 * half;
                C[(long)row * Ncols + (cg + 16 * t + l16)] = acc[t][r];
            }
        }
    }
}

// ---------------------------------------------------------------------------
// Zero-fill
// ---------------------------------------------------------------------------
__global__ void zero_f32(float* __restrict__ p, long n) {
    long i = (long)blockIdx.x * blockDim.x + threadIdx.x;
    if (i < n) p[i] = 0.0f;
}

// ---------------------------------------------------------------------------
// GATv2 edge pass 1: s_e = att . leaky_relu(xl[src] + xr[dst]); segment max.
// One wave per edge; lanes stride the feature dim (coalesced).
// Edges e >= E are self-loops (src = dst = e - E).
// ---------------------------------------------------------------------------
__global__ void edge_score(const int* __restrict__ ei, int E, int Etot,
                           const float* __restrict__ xl,
                           const float* __restrict__ xr,
                           const float* __restrict__ att, int D,
                           float* __restrict__ s, unsigned* __restrict__ smax)
{
    int wid  = (blockIdx.x * blockDim.x + threadIdx.x) >> 5;
    int lane = threadIdx.x & 31;
    if (wid >= Etot) return;
    int src = (wid < E) ? ei[wid]     : (wid - E);
    int dst = (wid < E) ? ei[E + wid] : (wid - E);
    float p = 0.f;
    for (int h = lane; h < D; h += 32) {
        float v = xl[(long)src * D + h] + xr[(long)dst * D + h];
        v = (v > 0.f) ? v : 0.2f * v;            // leaky_relu, slope 0.2
        p += att[h] * v;
    }
#pragma unroll
    for (int o = 16; o > 0; o >>= 1) p += __shfl_xor(p, o, 32);
    if (lane == 0) {
        s[wid] = p;
        atomicMax(&smax[dst], fkey(p));
    }
}

// ---------------------------------------------------------------------------
// GATv2 edge pass 2: ex = exp(s - smax[dst]); denom[dst] += ex. Thread/edge.
// ---------------------------------------------------------------------------
__global__ void edge_exp(const int* __restrict__ ei, int E, int Etot,
                         const float* __restrict__ s,
                         const unsigned* __restrict__ smax,
                         float* __restrict__ exb, float* __restrict__ den)
{
    int e = blockIdx.x * blockDim.x + threadIdx.x;
    if (e >= Etot) return;
    int dst = (e < E) ? ei[E + e] : (e - E);
    float ex = __expf(s[e] - funkey(smax[dst]));
    exb[e] = ex;
    atomAddF(&den[dst], ex);
}

// ---------------------------------------------------------------------------
// GATv2 edge pass 3: acc[dst] += (ex/denom[dst]) * xl[src]. Wave per edge.
// ---------------------------------------------------------------------------
__global__ void edge_acc(const int* __restrict__ ei, int E, int Etot,
                         const float* __restrict__ xl,
                         const float* __restrict__ exb,
                         const float* __restrict__ den, int D,
                         float* __restrict__ acc)
{
    int wid  = (blockIdx.x * blockDim.x + threadIdx.x) >> 5;
    int lane = threadIdx.x & 31;
    if (wid >= Etot) return;
    int src = (wid < E) ? ei[wid]     : (wid - E);
    int dst = (wid < E) ? ei[E + wid] : (wid - E);
    float a = exb[wid] / (den[dst] + 1e-16f);
    for (int h = lane; h < D; h += 32)
        atomAddF(&acc[(long)dst * D + h], xl[(long)src * D + h] * a);
}

// ---------------------------------------------------------------------------
// h = elu(acc + bias), in place. D is a power of two.
// ---------------------------------------------------------------------------
__global__ void elu_bias(float* __restrict__ acc, const float* __restrict__ bias,
                         long n, int D)
{
    long i = (long)blockIdx.x * blockDim.x + threadIdx.x;
    if (i >= n) return;
    float v = acc[i] + bias[i & (D - 1)];
    acc[i] = (v > 0.f) ? v : (__expf(v) - 1.f);
}

// ---------------------------------------------------------------------------
// Row softmax over K=32 with bias: Z[n] = softmax(zl[n] + b2). Wave per node.
// ---------------------------------------------------------------------------
__global__ void row_softmax32(const float* __restrict__ zl,
                              const float* __restrict__ b2,
                              float* __restrict__ Z, int n)
{
    int node = (blockIdx.x * blockDim.x + threadIdx.x) >> 5;
    int lane = threadIdx.x & 31;
    if (node >= n) return;
    float v = zl[(long)node * 32 + lane] + b2[lane];
    float m = v;
#pragma unroll
    for (int o = 16; o > 0; o >>= 1) m = fmaxf(m, __shfl_xor(m, o, 32));
    float ex = __expf(v - m);
    float sum = ex;
#pragma unroll
    for (int o = 16; o > 0; o >>= 1) sum += __shfl_xor(sum, o, 32);
    Z[(long)node * 32 + lane] = ex / sum;
}

// ---------------------------------------------------------------------------
// out2 = xp (elementwise init before contamination scatter)
// ---------------------------------------------------------------------------
__global__ void copy_f32(float* __restrict__ dst, const float* __restrict__ src, long n) {
    long i = (long)blockIdx.x * blockDim.x + threadIdx.x;
    if (i < n) dst[i] = src[i];
}

// ---------------------------------------------------------------------------
// Spatial contamination: out2[row] += sigmoid(ew) * xp[col]. Wave per edge.
// ---------------------------------------------------------------------------
__global__ void contam_scatter(const int* __restrict__ eis, int E,
                               const float* __restrict__ xp,
                               const float* __restrict__ ew,
                               float* __restrict__ out2)
{
    int wid  = (blockIdx.x * blockDim.x + threadIdx.x) >> 5;
    int lane = threadIdx.x & 31;
    if (wid >= E) return;
    int row = eis[wid];
    int col = eis[E + wid];
    float w = 1.f / (1.f + __expf(-ew[0]));
    for (int d = lane; d < F_IN; d += 32)
        atomAddF(&out2[(long)row * F_IN + d], w * xp[(long)col * F_IN + d]);
}

__global__ void write_w(const float* __restrict__ ew, float* __restrict__ out) {
    if (threadIdx.x == 0 && blockIdx.x == 0)
        out[0] = 1.f / (1.f + __expf(-ew[0]));
}

// ---------------------------------------------------------------------------
// Launch
// ---------------------------------------------------------------------------
extern "C" void kernel_launch(void* const* d_in, const int* in_sizes, int n_in,
                              void* d_out, int out_size, void* d_ws, size_t ws_size,
                              hipStream_t stream)
{
    const float* X    = (const float*)d_in[0];
    const int*   eif  = (const int*)  d_in[1];
    const int*   eis  = (const int*)  d_in[2];
    const float* Wl1  = (const float*)d_in[3];
    const float* Wr1  = (const float*)d_in[4];
    const float* att1 = (const float*)d_in[5];
    const float* b1   = (const float*)d_in[6];
    const float* Wl2  = (const float*)d_in[7];
    const float* Wr2  = (const float*)d_in[8];
    const float* att2 = (const float*)d_in[9];
    const float* b2   = (const float*)d_in[10];
    const float* Mmat = (const float*)d_in[11];
    const float* ew   = (const float*)d_in[12];

    const int N    = NN;
    const int Ef   = in_sizes[1] / 2;    // 800000 feature-graph edges
    const int Es   = in_sizes[2] / 2;    // 800000 spatial edges
    const int Etot = Ef + N;             // + self loops

    // ---------------- workspace layout (floats) ----------------
    float* ws = (float*)d_ws;
    float*    XL   = ws;                          // N*HDIM  (layer1 xl / layer2 xl)
    float*    XR   = XL  + (long)N * HDIM;        // N*HDIM
    float*    ACC  = XR  + (long)N * HDIM;        // N*HDIM  (layer1 acc -> h)
    float*    ZL   = ACC + (long)N * HDIM;        // N*KDIM  (layer2 acc)
    float*    S    = ZL  + (long)N * KDIM;        // Etot scores
    float*    EXB  = S   + (long)Etot;            // Etot exp()
    unsigned* SMAX = (unsigned*)(EXB + (long)Etot); // N keys
    float*    DEN  = (float*)(SMAX + N);          // N denom
    float*    XP   = DEN + N;                     // N*F_IN

    float* Zout = (float*)d_out;                  // [N, KDIM]
    float* Out2 = Zout + (long)N * KDIM;          // [N, F_IN]
    float* Wout = Out2 + (long)N * F_IN;          // scalar

    const int TB = 256;
    const int wavesPerBlk = TB / 32;
    dim3 blk(TB);
    auto gridE_wave  = dim3((Etot + wavesPerBlk - 1) / wavesPerBlk);
    auto gridEs_wave = dim3((Es   + wavesPerBlk - 1) / wavesPerBlk);
    auto gridN_wave  = dim3((N    + wavesPerBlk - 1) / wavesPerBlk);
    auto gridE_thr   = dim3((Etot + TB - 1) / TB);
    const int rowTiles  = N / 16;                 // exact: N % 16 == 0
    auto gridGemm = dim3((rowTiles + wavesPerBlk - 1) / wavesPerBlk);
    auto g1 = [&](long n) { return dim3((unsigned)((n + TB - 1) / TB)); };
    // Dynamic LDS for the B panel: Ncols * (Kd + 8) bf16 elements
    auto shB = [](int Kd, int Ncols) { return (size_t)Ncols * (Kd + 8) * 2; };

    // ================= Layer 1: GATv2(X; 256 -> 64) =================
    zero_f32<<<g1((long)N * HDIM), blk, 0, stream>>>(ACC, (long)N * HDIM);
    zero_f32<<<g1(N), blk, 0, stream>>>((float*)SMAX, N);
    zero_f32<<<g1(N), blk, 0, stream>>>(DEN, N);

    gemm_wmma<4, false><<<gridGemm, blk, shB(F_IN, HDIM), stream>>>(X, Wl1, XL, N, F_IN, HDIM);
    gemm_wmma<4, false><<<gridGemm, blk, shB(F_IN, HDIM), stream>>>(X, Wr1, XR, N, F_IN, HDIM);

    edge_score<<<gridE_wave, blk, 0, stream>>>(eif, Ef, Etot, XL, XR, att1, HDIM, S, SMAX);
    edge_exp  <<<gridE_thr,  blk, 0, stream>>>(eif, Ef, Etot, S, SMAX, EXB, DEN);
    edge_acc  <<<gridE_wave, blk, 0, stream>>>(eif, Ef, Etot, XL, EXB, DEN, HDIM, ACC);
    elu_bias  <<<g1((long)N * HDIM), blk, 0, stream>>>(ACC, b1, (long)N * HDIM, HDIM);

    // ================= Layer 2: GATv2(h; 64 -> 32) =================
    zero_f32<<<g1((long)N * KDIM), blk, 0, stream>>>(ZL, (long)N * KDIM);
    zero_f32<<<g1(N), blk, 0, stream>>>((float*)SMAX, N);
    zero_f32<<<g1(N), blk, 0, stream>>>(DEN, N);

    gemm_wmma<2, false><<<gridGemm, blk, shB(HDIM, KDIM), stream>>>(ACC, Wl2, XL, N, HDIM, KDIM);
    gemm_wmma<2, false><<<gridGemm, blk, shB(HDIM, KDIM), stream>>>(ACC, Wr2, XR, N, HDIM, KDIM);

    edge_score<<<gridE_wave, blk, 0, stream>>>(eif, Ef, Etot, XL, XR, att2, KDIM, S, SMAX);
    edge_exp  <<<gridE_thr,  blk, 0, stream>>>(eif, Ef, Etot, S, SMAX, EXB, DEN);
    edge_acc  <<<gridE_wave, blk, 0, stream>>>(eif, Ef, Etot, XL, EXB, DEN, KDIM, ZL);

    row_softmax32<<<gridN_wave, blk, 0, stream>>>(ZL, b2, Zout, N);

    // ================= X_pure = Z @ relu(M); contamination ============
    gemm_wmma<4, true><<<gridGemm, blk, shB(KDIM, F_IN), stream>>>(Zout, Mmat, XP, N, KDIM, F_IN);

    copy_f32<<<g1((long)N * F_IN), blk, 0, stream>>>(Out2, XP, (long)N * F_IN);
    contam_scatter<<<gridEs_wave, blk, 0, stream>>>(eis, Es, XP, ew, Out2);
    write_w<<<1, 32, 0, stream>>>(ew, Wout);

    (void)n_in; (void)out_size; (void)ws_size;
}
